// NodeFeatureExtractor_88493506167308
// MI455X (gfx1250) — matrix operand order repaired and checked by reading, not compiled, not run on hardware
//
#include <hip/hip_runtime.h>
#include <hip/hip_bf16.h>

typedef __attribute__((ext_vector_type(16))) __bf16 v16bf;
typedef __attribute__((ext_vector_type(8)))  __bf16 v8bf;
typedef __attribute__((ext_vector_type(8)))  float  v8f;
typedef __attribute__((ext_vector_type(4)))  float  v4f;

#define HID    128
#define NT     8      // 128 outputs / 16
#define KS1    16     // 512 (488 padded) / 32
#define KS2    4      // 128 / 32
#define FPIX   16384  // 128*128
#define INDIM  488
#define WAVES  12     // waves per block (3 per SIMD32 -> better gather-latency hiding)
#define BLKTHREADS (WAVES * 32)

// ---------------------------------------------------------------------------
// Degree kernels (segment_sum of ones over edge endpoints + max reduction)
// ---------------------------------------------------------------------------
__global__ void zero_deg_kernel(int* deg, int n) {
  int i = blockIdx.x * blockDim.x + threadIdx.x;
  if (i < n) deg[i] = 0;
}

__global__ void accum_deg_kernel(const int* __restrict__ ei, int* __restrict__ deg,
                                 int total, int NN) {
  int i = blockIdx.x * blockDim.x + threadIdx.x;
  if (i < total) {
    int v = ei[i];
    if (v >= 0 && v < NN) atomicAdd(&deg[v], 1);
  }
}

__global__ void max_deg_kernel(const int* __restrict__ deg, int* mx, int NN) {
  int stride = gridDim.x * blockDim.x;
  int best = 0;
  for (int i = blockIdx.x * blockDim.x + threadIdx.x; i < NN; i += stride)
    best = best > deg[i] ? best : deg[i];
  atomicMax(mx, best);
}

// ---------------------------------------------------------------------------
// Transpose [C][H][W] channel-major features into pixel-major [H*W][488]
// so bilinear gathers become contiguous b128 loads (L2-resident, 32MB).
// k slots: 0,1 zero | 2..481 backbone | 482..485 seg | 486,487 zero.
// ---------------------------------------------------------------------------
__global__ __launch_bounds__(256) void pack_feats_kernel(
    const float* __restrict__ bb, const float* __restrict__ seg,
    float* __restrict__ packed) {
  __shared__ float tile[64][65];
  int pix0 = blockIdx.x * 64;
  int k0   = blockIdx.y * 64;
  int tid  = threadIdx.x;
  int px = tid & 63, kk = tid >> 6;          // 4 k-rows per pass
  #pragma unroll
  for (int r = 0; r < 16; ++r) {
    int k = k0 + kk + r * 4;
    float v = 0.0f;
    if (k >= 2 && k < 482)        v = bb[(size_t)(k - 2) * FPIX + pix0 + px];
    else if (k >= 482 && k < 486) v = seg[(size_t)(k - 482) * FPIX + pix0 + px];
    tile[kk + r * 4][px] = v;
  }
  __syncthreads();
  int kk2 = tid & 63, px2 = tid >> 6;
  #pragma unroll
  for (int r = 0; r < 16; ++r) {
    int k = k0 + kk2;
    int p = pix0 + px2 + r * 4;
    if (k < INDIM) packed[(size_t)p * INDIM + k] = tile[kk2][px2 + r * 4];
  }
}

// ---------------------------------------------------------------------------
// Build 8 consecutive input features starting at K0 for one node.
// Packed path: two v4f loads per corner (32B-aligned since K0 % 8 == 0).
// ---------------------------------------------------------------------------
__device__ __forceinline__ void feat_run8(
    float* f, int K0, int usePacked,
    const float* __restrict__ packed,
    const float* __restrict__ bb, const float* __restrict__ seg,
    int p00, int p01, int p10, int p11,
    float w00, float w01, float w10, float w11) {
  if (K0 >= INDIM) {
    #pragma unroll
    for (int j = 0; j < 8; ++j) f[j] = 0.0f;
    return;
  }
  if (usePacked) {
    const float* b00 = packed + (size_t)p00 * INDIM + K0;
    const float* b01 = packed + (size_t)p01 * INDIM + K0;
    const float* b10 = packed + (size_t)p10 * INDIM + K0;
    const float* b11 = packed + (size_t)p11 * INDIM + K0;
    #pragma unroll
    for (int h = 0; h < 2; ++h) {
      v4f c00 = *(const v4f*)(b00 + 4 * h);
      v4f c01 = *(const v4f*)(b01 + 4 * h);
      v4f c10 = *(const v4f*)(b10 + 4 * h);
      v4f c11 = *(const v4f*)(b11 + 4 * h);
      #pragma unroll
      for (int k = 0; k < 4; ++k)
        f[4 * h + k] = c00[k] * w00 + c01[k] * w01 + c10[k] * w10 + c11[k] * w11;
    }
  } else {
    #pragma unroll
    for (int j = 0; j < 8; ++j) {
      int K = K0 + j;
      float v = 0.0f;
      if (K >= 2 && K < 486) {
        int c = K - 2;
        const float* p = (c < 480) ? (bb + (size_t)c * FPIX)
                                   : (seg + (size_t)(c - 480) * FPIX);
        v = p[p00] * w00 + p[p01] * w01 + p[p10] * w10 + p[p11] * w11;
      }
      f[j] = v;
    }
  }
}

// ---------------------------------------------------------------------------
// Main fused kernel: bilinear sample + feature assembly + 2-layer MLP on WMMA.
// One wave (32 lanes) = one 16-node tile; block = 12 waves = 192 nodes.
// ---------------------------------------------------------------------------
__global__ __launch_bounds__(BLKTHREADS) void node_mlp_kernel(
    const float* __restrict__ vertices,
    const float* __restrict__ bb,
    const float* __restrict__ seg,
    const float* __restrict__ packed,
    const float* __restrict__ W1, const float* __restrict__ b1,
    const float* __restrict__ W2, const float* __restrict__ b2,
    const int*   __restrict__ deg,
    const int*   __restrict__ imsz,
    float* __restrict__ out,
    int NN, int usePacked) {
  // Weights pre-swizzled to bf16 WMMA B-fragment order; per-wave hidden tiles.
  __shared__ __align__(32) __bf16 sW1[NT * KS1 * 32 * 16];  // 128 KB
  __shared__ __align__(32) __bf16 sW2[NT * KS2 * 32 * 16];  //  32 KB
  __shared__ __align__(32) __bf16 sH[WAVES][16][HID];       //  48 KB

  const int tid = threadIdx.x;

  // ---- stage W1/W2 into LDS (bf16, B-fragment layout) -------------------
  for (int idx = tid; idx < NT * KS1 * 32 * 16; idx += BLKTHREADS) {
    int nt = idx >> 13, r = idx & 8191;
    int s = r >> 9;   int r2 = r & 511;
    int ln = r2 >> 4; int j = r2 & 15;
    int N = nt * 16 + (ln & 15);
    int K = 32 * s + (j < 8 ? j : j + 8) + ((ln & 16) ? 8 : 0);
    float v = (K < INDIM) ? W1[(size_t)N * INDIM + K] : 0.0f;
    sW1[idx] = (__bf16)v;
  }
  for (int idx = tid; idx < NT * KS2 * 32 * 16; idx += BLKTHREADS) {
    int nt = idx >> 11, r = idx & 2047;
    int s = r >> 9;   int r2 = r & 511;
    int ln = r2 >> 4; int j = r2 & 15;
    int N = nt * 16 + (ln & 15);
    int K = 32 * s + (j < 8 ? j : j + 8) + ((ln & 16) ? 8 : 0);
    sW2[idx] = (__bf16)W2[(size_t)N * HID + K];
  }
  __syncthreads();

  // ---- per-lane node setup ---------------------------------------------
  const int w    = tid >> 5;
  const int lane = tid & 31;
  const int m    = lane & 15;
  const int koff = (lane & 16) ? 8 : 0;   // K-half split of A layout

  long tile = (long)blockIdx.x * WAVES + w;
  int base = (int)(tile * 16);
  int node = base + m;
  node = node < NN ? node : NN - 1;
  node = node > 0 ? node : 0;

  float S  = (float)imsz[0];
  float vx = vertices[(size_t)node * 2 + 0];
  float vy = vertices[(size_t)node * 2 + 1];
  float ix = vx / S * 127.0f, iy = vy / S * 127.0f;
  float x0 = floorf(ix), y0 = floorf(iy);
  float wx = ix - x0, wy = iy - y0;
  int x0i = (int)x0; x0i = x0i < 0 ? 0 : (x0i > 127 ? 127 : x0i);
  int y0i = (int)y0; y0i = y0i < 0 ? 0 : (y0i > 127 ? 127 : y0i);
  int x1i = x0i + 1 > 127 ? 127 : x0i + 1;
  int y1i = y0i + 1 > 127 ? 127 : y0i + 1;
  float w00 = (1.0f - wx) * (1.0f - wy), w01 = wx * (1.0f - wy);
  float w10 = (1.0f - wx) * wy,          w11 = wx * wy;
  int p00 = y0i * 128 + x0i, p01 = y0i * 128 + x1i;
  int p10 = y1i * 128 + x0i, p11 = y1i * 128 + x1i;

  float maxd   = (float)deg[NN];
  float dnorm  = (float)deg[node] / (maxd + 1e-6f);
  float dxb    = fminf(vx, S - vx), dyb = fminf(vy, S - vy);
  float dbound = fminf(dxb, dyb) / (S * 0.5f);
  float cnx = vx / S, cny = vy / S;

  // ---- GEMM1: features(16x488) @ W1^T -> h(16x128), bias in accumulators
  v8f acc[NT];
  #pragma unroll
  for (int t = 0; t < NT; ++t) {
    float bias = b1[t * 16 + m];
    #pragma unroll
    for (int i = 0; i < 8; ++i) acc[t][i] = bias;
  }

  for (int s = 0; s < KS1; ++s) {
    float f[16];
    feat_run8(f,     32 * s + koff,      usePacked, packed, bb, seg,
              p00, p01, p10, p11, w00, w01, w10, w11);
    feat_run8(f + 8, 32 * s + 16 + koff, usePacked, packed, bb, seg,
              p00, p01, p10, p11, w00, w01, w10, w11);
    if (s == 0 || s == KS1 - 1) {       // patch non-gather features
      #pragma unroll
      for (int j = 0; j < 16; ++j) {
        int K = 32 * s + (j < 8 ? j : j + 8) + koff;
        if (K == 0)        f[j] = cnx;
        else if (K == 1)   f[j] = cny;
        else if (K == 486) f[j] = dnorm;
        else if (K == 487) f[j] = dbound;
      }
    }
    v16bf a;
    #pragma unroll
    for (int j = 0; j < 16; ++j) a[j] = (__bf16)f[j];
    #pragma unroll
    for (int t = 0; t < NT; ++t) {
      v16bf bfrag = *(const v16bf*)&sW1[((t * KS1 + s) * 32 + lane) * 16];
      acc[t] = __builtin_amdgcn_wmma_f32_16x16x32_bf16(
          false, a, false, bfrag, (short)0, acc[t], false, false);
    }
  }

  // ---- ReLU, spill hidden tile to LDS (transpose for GEMM2 A-fragments)
  const int rOff = koff;  // row offset 0 / 8 for D-layout
  #pragma unroll
  for (int t = 0; t < NT; ++t)
    #pragma unroll
    for (int i = 0; i < 8; ++i)
      sH[w][i + rOff][t * 16 + m] = (__bf16)fmaxf(acc[t][i], 0.0f);
  __syncthreads();  // uniform across block; orders DS write->read cross-lane

  // ---- GEMM2: h(16x128) @ W2^T -> out(16x128) ---------------------------
  v8f acc2[NT];
  #pragma unroll
  for (int t = 0; t < NT; ++t) {
    float bias = b2[t * 16 + m];
    #pragma unroll
    for (int i = 0; i < 8; ++i) acc2[t][i] = bias;
  }
  const __bf16* hrow = &sH[w][m][0];
  #pragma unroll
  for (int s = 0; s < KS2; ++s) {
    int kb = 32 * s + koff;
    v8bf lo = *(const v8bf*)&hrow[kb];
    v8bf hi = *(const v8bf*)&hrow[kb + 16];
    v16bf a;
    #pragma unroll
    for (int j = 0; j < 8; ++j) { a[j] = lo[j]; a[j + 8] = hi[j]; }
    #pragma unroll
    for (int t = 0; t < NT; ++t) {
      v16bf bfrag = *(const v16bf*)&sW2[((t * KS2 + s) * 32 + lane) * 16];
      acc2[t] = __builtin_amdgcn_wmma_f32_16x16x32_bf16(
          false, a, false, bfrag, (short)0, acc2[t], false, false);
    }
  }

  // ---- ReLU + store -----------------------------------------------------
  #pragma unroll
  for (int t = 0; t < NT; ++t)
    #pragma unroll
    for (int i = 0; i < 8; ++i) {
      int nd = base + i + rOff;
      if (nd < NN) out[(size_t)nd * HID + t * 16 + m] = fmaxf(acc2[t][i], 0.0f);
    }
}

// ---------------------------------------------------------------------------
extern "C" void kernel_launch(void* const* d_in, const int* in_sizes, int n_in,
                              void* d_out, int out_size, void* d_ws, size_t ws_size,
                              hipStream_t stream) {
  const float* vertices = (const float*)d_in[0];
  const float* bb       = (const float*)d_in[1];
  const float* seg      = (const float*)d_in[2];
  const int*   ei       = (const int*)d_in[3];
  const float* W1       = (const float*)d_in[4];
  const float* b1       = (const float*)d_in[5];
  const float* W2       = (const float*)d_in[6];
  const float* b2       = (const float*)d_in[7];
  const int*   imsz     = (const int*)d_in[8];
  float* out = (float*)d_out;

  int NN       = in_sizes[0] / 2;   // nodes
  int totalIdx = in_sizes[3];       // 2*E endpoints

  int* deg = (int*)d_ws;            // [NN] degrees + [1] max
  size_t packOff   = (((size_t)(NN + 1) * 4) + 255) & ~(size_t)255;
  size_t packBytes = (size_t)FPIX * INDIM * 4;
  int usePacked = (ws_size >= packOff + packBytes) ? 1 : 0;
  float* packed = (float*)((char*)d_ws + packOff);

  zero_deg_kernel<<<(NN + 1 + 255) / 256, 256, 0, stream>>>(deg, NN + 1);
  accum_deg_kernel<<<(totalIdx + 255) / 256, 256, 0, stream>>>(ei, deg, totalIdx, NN);
  max_deg_kernel<<<256, 256, 0, stream>>>(deg, deg + NN, NN);
  if (usePacked)
    pack_feats_kernel<<<dim3(FPIX / 64, 8), 256, 0, stream>>>(bb, seg, packed);

  int numTiles = (NN + 15) / 16;
  int blocks   = (numTiles + WAVES - 1) / WAVES;
  node_mlp_kernel<<<blocks, BLKTHREADS, 0, stream>>>(
      vertices, bb, seg, packed, W1, b1, W2, b2, deg, imsz, out, NN, usePacked);
}
